// BidirectionalRNN_352187319218
// MI455X (gfx1250) — matrix operand order repaired and checked by reading, not compiled, not run on hardware
//
#include <hip/hip_runtime.h>
#include <hip/hip_bf16.h>
#include <math.h>

// Problem constants (match reference)
constexpr int kB = 64;
constexpr int kT = 512;
constexpr int kI = 512;
constexpr int kH = 512;

typedef __attribute__((ext_vector_type(16))) __bf16 v16bf;
typedef __attribute__((ext_vector_type(8)))  __bf16 v8bf;
typedef __attribute__((ext_vector_type(8)))  float  v8f;

// ---------------------------------------------------------------------------
// Fragment loaders (CDNA5 wave32 WMMA 16x16x32 bf16 layouts, ISA 7.12.2)
// A (16xK, row-major, ld): lane L holds row (L&15); k groups:
//   elems 0..7  -> k0 + 8*(L>>4) + e
//   elems 8..15 -> k0 + 16 + 8*(L>>4) + (e-8)
// B (KxN) taken from W[n][k] row-major (ld = K stride): lane L holds col
//   n0+(L&15); same k groups as A (elems 0..7 -> k0+8*(L>>4)+e, 8..15 -> +16)
// ---------------------------------------------------------------------------
__device__ __forceinline__ v16bf load_frag_a(const __bf16* __restrict__ base,
                                             int ld, int k0, int lane) {
  const int row  = lane & 15;
  const int half = lane >> 4;
  const __bf16* p = base + (size_t)row * ld + k0 + 8 * half;
  union { v16bf v; v8bf h[2]; } u;
  u.h[0] = *(const v8bf*)(p);
  u.h[1] = *(const v8bf*)(p + 16);
  return u.v;
}

__device__ __forceinline__ v16bf load_frag_b(const __bf16* __restrict__ w,
                                             int ld, int n0, int k0, int lane) {
  const int col  = n0 + (lane & 15);
  const int half = lane >> 4;
  const __bf16* p = w + (size_t)col * ld + k0 + 8 * half;
  union { v16bf v; v8bf h[2]; } u;
  u.h[0] = *(const v8bf*)(p);
  u.h[1] = *(const v8bf*)(p + 16);
  return u.v;
}

__device__ __forceinline__ v8f wmma_bf16(v16bf a, v16bf b, v8f c) {
  return __builtin_amdgcn_wmma_f32_16x16x32_bf16(false, a, false, b,
                                                 (short)0, c, false, false);
}

// ---------------------------------------------------------------------------
// Phase 0: fp32 -> bf16 conversion (grid-stride)
// ---------------------------------------------------------------------------
__global__ void cvt_f32_to_bf16(const float* __restrict__ src,
                                __bf16* __restrict__ dst, int n) {
  int i = blockIdx.x * blockDim.x + threadIdx.x;
  int stride = gridDim.x * blockDim.x;
  for (; i < n; i += stride) dst[i] = (__bf16)src[i];
}

// ---------------------------------------------------------------------------
// Phase 1: xproj[dir][t][b][h] = bf16( x[b,t,:] . Wx_dir[h,:] + bias_dir[h] )
// One wave owns a 16-row strip of A (flat m = b*T + t), A resident in VGPRs
// (16 fragments = 128 VGPRs), loops all 32 N-tiles streaming B from L2.
// grid = (strips/8, 2 dirs), block = 256 (8 waves).
// ---------------------------------------------------------------------------
__global__ __launch_bounds__(256)
void rnn_xproj(const __bf16* __restrict__ xbf,
               const __bf16* __restrict__ wxbf,
               const float* __restrict__ bias_f,
               const float* __restrict__ bias_b,
               __bf16* __restrict__ xproj) {
  const int dir  = blockIdx.y;
  const int wave = threadIdx.x >> 5;
  const int lane = threadIdx.x & 31;
  const int strip = blockIdx.x * 8 + wave;       // 2048 strips of 16 rows
  const int m0 = strip * 16;

  const __bf16* __restrict__ wx = wxbf + (size_t)dir * kH * kI;
  const float*  __restrict__ bias = dir ? bias_b : bias_f;
  const __bf16* __restrict__ abase = xbf + (size_t)m0 * kI;

  // Preload A strip: 16 k-chunks -> 128 VGPRs
  v16bf afr[16];
#pragma unroll
  for (int kc = 0; kc < 16; ++kc)
    afr[kc] = load_frag_a(abase, kI, kc * 32, lane);

  const int half = lane >> 4;
  const v8f zero = {0.f, 0.f, 0.f, 0.f, 0.f, 0.f, 0.f, 0.f};

  for (int nt = 0; nt < 32; ++nt) {
    const int n0 = nt * 16;
    v8f acc = zero;
#pragma unroll
    for (int kc = 0; kc < 16; ++kc) {
      v16bf bf = load_frag_b(wx, kI, n0, kc * 32, lane);
      acc = wmma_bf16(afr[kc], bf, acc);
    }
    const int n = n0 + (lane & 15);
    const float bv = bias[n];
#pragma unroll
    for (int r = 0; r < 8; ++r) {
      const int m  = m0 + 8 * half + r;
      const int t  = m & (kT - 1);
      const int bb = m >> 9;                      // m / T
      xproj[(((size_t)dir * kT + t) * kB + bb) * kH + n] =
          (__bf16)(acc[r] + bv);
    }
  }
}

// ---------------------------------------------------------------------------
// Phase 2: sequential scan. grid = (4 batch-groups, 2 dirs), block = 256.
// Workgroup keeps h (16 x 512, bf16) in LDS; per step each wave computes a
// 16x64 slice of h_new = tanh(xproj_t + h @ Wh^T) with 64 WMMAs.
// Wh (512 KB bf16) streams from L2 each step; y stored fp32 to d_out.
// ---------------------------------------------------------------------------
constexpr int kHLD = kH + 8;                      // LDS row pad: 16B shift/row

__global__ __launch_bounds__(256)
void rnn_scan(const __bf16* __restrict__ whbf,
              const __bf16* __restrict__ xproj,
              const float* __restrict__ h0_f,
              const float* __restrict__ h0_b,
              float* __restrict__ out) {
  const int dir = blockIdx.y;
  const int bg  = blockIdx.x;
  const int b0  = bg * 16;

  __shared__ __bf16 hsh[16 * kHLD];               // 16.6 KB of the 320 KB LDS

  // Init h from h0 (fp32 -> bf16)
  const float* __restrict__ h0 = dir ? h0_b : h0_f;
  for (int i = threadIdx.x; i < 16 * kH; i += 256) {
    const int r = i >> 9;
    const int c = i & (kH - 1);
    hsh[r * kHLD + c] = (__bf16)h0[(size_t)(b0 + r) * kH + c];
  }
  __syncthreads();

  const int wave = threadIdx.x >> 5;
  const int lane = threadIdx.x & 31;
  const int half = lane >> 4;
  const int n0w  = wave * 64;

  const __bf16* __restrict__ wh = whbf + (size_t)dir * kH * kH;
  float* __restrict__ hT = out + (size_t)kB * kT * 2 * kH + (size_t)dir * kB * kH;
  const v8f zero = {0.f, 0.f, 0.f, 0.f, 0.f, 0.f, 0.f, 0.f};

  for (int s = 0; s < kT; ++s) {
    const int t = dir ? (kT - 1 - s) : s;

    v8f acc0 = zero, acc1 = zero, acc2 = zero, acc3 = zero;
#pragma unroll 4
    for (int kc = 0; kc < 16; ++kc) {
      const int k0 = kc * 32;
      // A fragment from LDS h (reused across the 4 N-tiles)
      const __bf16* ap = &hsh[(lane & 15) * kHLD + k0 + 8 * half];
      union { v16bf v; v8bf h[2]; } ua;
      ua.h[0] = *(const v8bf*)(ap);
      ua.h[1] = *(const v8bf*)(ap + 16);
      acc0 = wmma_bf16(ua.v, load_frag_b(wh, kH, n0w +  0, k0, lane), acc0);
      acc1 = wmma_bf16(ua.v, load_frag_b(wh, kH, n0w + 16, k0, lane), acc1);
      acc2 = wmma_bf16(ua.v, load_frag_b(wh, kH, n0w + 32, k0, lane), acc2);
      acc3 = wmma_bf16(ua.v, load_frag_b(wh, kH, n0w + 48, k0, lane), acc3);
    }

    // xproj add + tanh + y store + prepare h_new
    __bf16 hnew[4][8];
    v8f accs[4] = {acc0, acc1, acc2, acc3};
#pragma unroll
    for (int nt = 0; nt < 4; ++nt) {
      const int n = n0w + nt * 16 + (lane & 15);
#pragma unroll
      for (int r = 0; r < 8; ++r) {
        const int m  = 8 * half + r;
        const int bb = b0 + m;
        float v = accs[nt][r] +
                  (float)xproj[(((size_t)dir * kT + t) * kB + bb) * kH + n];
        v = tanhf(v);
        out[((size_t)bb * kT + t) * (2 * kH) + (size_t)dir * kH + n] = v;
        hnew[nt][r] = (__bf16)v;
        if (s == kT - 1) hT[(size_t)bb * kH + n] = v;
      }
    }

    __syncthreads();   // everyone done reading old h
#pragma unroll
    for (int nt = 0; nt < 4; ++nt) {
      const int n = n0w + nt * 16 + (lane & 15);
#pragma unroll
      for (int r = 0; r < 8; ++r)
        hsh[(8 * half + r) * kHLD + n] = hnew[nt][r];
    }
    __syncthreads();   // h fully updated for next step
  }
}

// ---------------------------------------------------------------------------
// Host launcher
// ---------------------------------------------------------------------------
extern "C" void kernel_launch(void* const* d_in, const int* in_sizes, int n_in,
                              void* d_out, int out_size, void* d_ws, size_t ws_size,
                              hipStream_t stream) {
  const float* x   = (const float*)d_in[0];
  const float* h0f = (const float*)d_in[1];
  const float* h0b = (const float*)d_in[2];
  const float* Wxf = (const float*)d_in[3];
  const float* bxf = (const float*)d_in[4];
  const float* Whf = (const float*)d_in[5];
  const float* Wxb = (const float*)d_in[6];
  const float* bxb = (const float*)d_in[7];
  const float* Whb = (const float*)d_in[8];

  // Workspace layout (bf16): x | Wx(2) | Wh(2) | xproj(2)  ~= 98 MB
  __bf16* xbf   = (__bf16*)d_ws;
  __bf16* wxbf  = xbf  + (size_t)kB * kT * kI;
  __bf16* whbf  = wxbf + 2 * (size_t)kH * kI;
  __bf16* xproj = whbf + 2 * (size_t)kH * kH;

  cvt_f32_to_bf16<<<4096, 256, 0, stream>>>(x,   xbf,            kB * kT * kI);
  cvt_f32_to_bf16<<<512,  256, 0, stream>>>(Wxf, wxbf,           kH * kI);
  cvt_f32_to_bf16<<<512,  256, 0, stream>>>(Wxb, wxbf + (size_t)kH * kI, kH * kI);
  cvt_f32_to_bf16<<<512,  256, 0, stream>>>(Whf, whbf,           kH * kH);
  cvt_f32_to_bf16<<<512,  256, 0, stream>>>(Whb, whbf + (size_t)kH * kH, kH * kH);

  dim3 g1(256, 2);     // 2048 strips / 8 waves-per-block, x 2 directions
  rnn_xproj<<<g1, 256, 0, stream>>>(xbf, wxbf, bxf, bxb, xproj);

  dim3 g2(4, 2);       // 4 batch-groups x 2 directions
  rnn_scan<<<g2, 256, 0, stream>>>(whbf, xproj, h0f, h0b, (float*)d_out);
}
